// EncoderDecoderLSTM_66408784331428
// MI455X (gfx1250) — matrix operand order repaired and checked by reading, not compiled, not run on hardware
//
#include <hip/hip_runtime.h>
#include <hip/hip_bf16.h>
#include <math.h>

typedef __bf16 bf16;
typedef __attribute__((ext_vector_type(16))) __bf16 v16bf;
typedef __attribute__((ext_vector_type(8)))  __bf16 v8bf;
typedef __attribute__((ext_vector_type(8)))  float  v8f;

#define TILE_LD 72   // bf16 elems per LDS row: 64 data + 8 pad (144B row keeps 16B alignment)

// ---------- helpers ----------

__device__ __forceinline__ bf16 f2bf(float f) {
    union { float f; unsigned u; } v; v.f = f;
    unsigned r = v.u + 0x7FFFu + ((v.u >> 16) & 1u);   // round-to-nearest-even
    unsigned short s = (unsigned short)(r >> 16);
    return __builtin_bit_cast(bf16, s);
}

__device__ __forceinline__ v16bf join16(v8bf lo, v8bf hi) {
    return __builtin_shufflevector(lo, hi, 0,1,2,3,4,5,6,7,8,9,10,11,12,13,14,15);
}

__device__ __forceinline__ float sigm(float x) { return 1.0f / (1.0f + expf(-x)); }

// register-staged cooperative load of a 64x64 bf16 tile (256 threads, 16 elems each)
__device__ __forceinline__ void preload_tile(const bf16* __restrict__ P, int ld,
                                             int r0, int k0, v8bf& lo, v8bf& hi) {
    int tid = threadIdx.x;
    int row = tid >> 2;
    int kq  = (tid & 3) << 4;            // 0,16,32,48
    const bf16* p = P + (size_t)(r0 + row) * ld + (k0 + kq);
    lo = *(const v8bf*)p;
    hi = *(const v8bf*)(p + 8);
}

__device__ __forceinline__ void stage_tile(bf16* S, v8bf lo, v8bf hi) {
    int tid = threadIdx.x;
    int row = tid >> 2;
    int kq  = (tid & 3) << 4;
    *(v8bf*)&S[row * TILE_LD + kq]     = lo;
    *(v8bf*)&S[row * TILE_LD + kq + 8] = hi;
}

// one 16x16x32 WMMA at K-offset ko; fragment gathers per CDNA5 16-bit A/B striping tables
__device__ __forceinline__ void wave_mma(const bf16* Xs, const bf16* Ws,
                                         int tm, int tn, int ko, v8f& acc) {
    int lane = threadIdx.x & 31;
    int half = lane >> 4;
    int l    = lane & 15;
    // A (16x32): lane group g: v0..3 = K g*8..+7 ; v4..7 = K 16+g*8..+7
    v8bf a_lo = *(const v8bf*)&Xs[(tm*16 + l) * TILE_LD + ko + half*8];
    v8bf a_hi = *(const v8bf*)&Xs[(tm*16 + l) * TILE_LD + ko + 16 + half*8];
    // B (32x16): lane group g: v0..3 = K g*8..+7 ; v4..7 = K 16+g*8..+7
    v8bf b_lo = *(const v8bf*)&Ws[(tn*16 + l) * TILE_LD + ko + half*8];
    v8bf b_hi = *(const v8bf*)&Ws[(tn*16 + l) * TILE_LD + ko + 16 + half*8];
    acc = __builtin_amdgcn_wmma_f32_16x16x32_bf16(false, join16(a_lo, a_hi),
                                                  false, join16(b_lo, b_hi),
                                                  (short)0, acc, false, false);
}

// ---------- kernels ----------

__global__ __launch_bounds__(256) void k_cvt_bf16(const float* __restrict__ s,
                                                  bf16* __restrict__ d, int n) {
    int i = blockIdx.x * 256 + threadIdx.x;
    if (i < n) d[i] = f2bf(s[i]);
}

__global__ __launch_bounds__(256) void k_zero32(unsigned* __restrict__ p, int n) {
    int i = blockIdx.x * 256 + threadIdx.x;
    if (i < n) p[i] = 0u;
}

// Y = relu(X[512,K] @ W^T + bias); bf16 in, bf16 out (+ optional fp32 copy, used by last layer)
// 64x64 block, 8 waves, 2 tiles/wave, K-stage 64, double-buffered LDS (one barrier per stage)
__global__ __launch_bounds__(256)
void k_gemm_bias_relu(const bf16* __restrict__ X, int ldx,
                      const bf16* __restrict__ W, int K,
                      const float* __restrict__ bias,
                      bf16* __restrict__ Yb, int ldyb,
                      float* __restrict__ Yf, int ldyf) {
    __shared__ bf16 Xs[2][64 * TILE_LD];
    __shared__ bf16 Ws[2][64 * TILE_LD];
    const int n0 = blockIdx.x * 64;
    const int m0 = blockIdx.y * 64;
    const int wid = threadIdx.x >> 5;
    const int tn  = wid & 3;
    const int tm  = wid >> 2;          // tiles tm and tm+2 share the B fragment
    v8f acc0 = {}; v8f acc1 = {};
    v8bf x0, x1, w0, w1;
    preload_tile(X, ldx, m0, 0, x0, x1);
    preload_tile(W, K,  n0, 0, w0, w1);
    int s = 0;
    for (int k0 = 0; k0 < K; k0 += 64, ++s) {
        bf16* Xc = Xs[s & 1];
        bf16* Wc = Ws[s & 1];
        stage_tile(Xc, x0, x1);
        stage_tile(Wc, w0, w1);
        __syncthreads();                                    // single barrier per stage
        if (k0 + 64 < K) {                                  // overlap next global loads
            preload_tile(X, ldx, m0, k0 + 64, x0, x1);
            preload_tile(W, K,  n0, k0 + 64, w0, w1);
        }
        wave_mma(Xc, Wc, tm,     tn, 0,  acc0);
        wave_mma(Xc, Wc, tm + 2, tn, 0,  acc1);
        wave_mma(Xc, Wc, tm,     tn, 32, acc0);
        wave_mma(Xc, Wc, tm + 2, tn, 32, acc1);
    }
    const int lane = threadIdx.x & 31;
    const int half = lane >> 4;
    const int l    = lane & 15;
    const int col  = n0 + tn*16 + l;
    const float b  = bias[col];
#pragma unroll
    for (int r = 0; r < 8; ++r) {
        int row0 = m0 + tm*16     + half*8 + r;
        int row1 = m0 + (tm+2)*16 + half*8 + r;
        float v0 = fmaxf(acc0[r] + b, 0.0f);
        float v1 = fmaxf(acc1[r] + b, 0.0f);
        Yb[(size_t)row0 * ldyb + col] = f2bf(v0);
        Yb[(size_t)row1 * ldyb + col] = f2bf(v1);
        if (Yf) {
            Yf[(size_t)row0 * ldyf + col] = v0;
            Yf[(size_t)row1 * ldyf + col] = v1;
        }
    }
}

// gates[512,4096] = X[512,256] @ Wih^T + H[512,1024] @ Whh^T + (bih + bhh)   (fp32 out)
__global__ __launch_bounds__(256)
void k_gates(const bf16* __restrict__ X, int ldx,
             const bf16* __restrict__ H,
             const bf16* __restrict__ Wih,
             const bf16* __restrict__ Whh,
             const float* __restrict__ bih,
             const float* __restrict__ bhh,
             float* __restrict__ G) {
    __shared__ bf16 Xs[2][64 * TILE_LD];
    __shared__ bf16 Ws[2][64 * TILE_LD];
    const int n0 = blockIdx.x * 64;
    const int m0 = blockIdx.y * 64;
    const int wid = threadIdx.x >> 5;
    const int tn  = wid & 3;
    const int tm  = wid >> 2;
    v8f acc0 = {}; v8f acc1 = {};
    v8bf x0, x1, w0, w1;
    int s = 0;
    // phase 1: input contribution, K = 256
    preload_tile(X, ldx, m0, 0, x0, x1);
    preload_tile(Wih, 256, n0, 0, w0, w1);
    for (int k0 = 0; k0 < 256; k0 += 64, ++s) {
        bf16* Xc = Xs[s & 1];
        bf16* Wc = Ws[s & 1];
        stage_tile(Xc, x0, x1);
        stage_tile(Wc, w0, w1);
        __syncthreads();
        if (k0 + 64 < 256) {
            preload_tile(X, ldx, m0, k0 + 64, x0, x1);
            preload_tile(Wih, 256, n0, k0 + 64, w0, w1);
        } else {                                            // prime phase 2
            preload_tile(H, 1024, m0, 0, x0, x1);
            preload_tile(Whh, 1024, n0, 0, w0, w1);
        }
        wave_mma(Xc, Wc, tm,     tn, 0,  acc0);
        wave_mma(Xc, Wc, tm + 2, tn, 0,  acc1);
        wave_mma(Xc, Wc, tm,     tn, 32, acc0);
        wave_mma(Xc, Wc, tm + 2, tn, 32, acc1);
    }
    // phase 2: recurrent contribution, K = 1024
    for (int k0 = 0; k0 < 1024; k0 += 64, ++s) {
        bf16* Xc = Xs[s & 1];
        bf16* Wc = Ws[s & 1];
        stage_tile(Xc, x0, x1);
        stage_tile(Wc, w0, w1);
        __syncthreads();
        if (k0 + 64 < 1024) {
            preload_tile(H, 1024, m0, k0 + 64, x0, x1);
            preload_tile(Whh, 1024, n0, k0 + 64, w0, w1);
        }
        wave_mma(Xc, Wc, tm,     tn, 0,  acc0);
        wave_mma(Xc, Wc, tm + 2, tn, 0,  acc1);
        wave_mma(Xc, Wc, tm,     tn, 32, acc0);
        wave_mma(Xc, Wc, tm + 2, tn, 32, acc1);
    }
    const int lane = threadIdx.x & 31;
    const int half = lane >> 4;
    const int l    = lane & 15;
    const int col  = n0 + tn*16 + l;
    const float b  = bih[col] + bhh[col];
#pragma unroll
    for (int r = 0; r < 8; ++r) {
        int row0 = m0 + tm*16     + half*8 + r;
        int row1 = m0 + (tm+2)*16 + half*8 + r;
        G[(size_t)row0 * 4096 + col] = acc0[r] + b;
        G[(size_t)row1 * 4096 + col] = acc1[r] + b;
    }
}

// LSTM pointwise: gate order i,f,g,o; c' = sig(f)*c + sig(i)*tanh(g); h' = sig(o)*tanh(c')
// c kept fp32, h emitted bf16 (GEMM consumers are bf16)
__global__ __launch_bounds__(256)
void k_lstm_pointwise(const float* __restrict__ G,
                      bf16* __restrict__ h, float* __restrict__ c) {
    int idx = blockIdx.x * 256 + threadIdx.x;     // 512*1024 threads
    int b = idx >> 10;
    int j = idx & 1023;
    const float* g = G + (size_t)b * 4096;
    float gi = sigm(g[j]);
    float gf = sigm(g[j + 1024]);
    float gg = tanhf(g[j + 2048]);
    float go = sigm(g[j + 3072]);
    float cn = gf * c[idx] + gi * gg;
    c[idx] = cn;
    h[idx] = f2bf(go * tanhf(cn));
}

// ---------- launch ----------

extern "C" void kernel_launch(void* const* d_in, const int* in_sizes, int n_in,
                              void* d_out, int out_size, void* d_ws, size_t ws_size,
                              hipStream_t stream) {
    (void)in_sizes; (void)n_in; (void)out_size; (void)ws_size;
    const int B = 512, PAST = 128, FUTURE = 64, T = PAST + FUTURE;
    const int I = 256, Hn = 1024, G4 = 4096, D0 = 1024, D1 = 512, OP = 256;

    const float* input_seq = (const float*)d_in[0];
    /* d_in[1] = future_n (fixed 64) */
    const float* W_ih = (const float*)d_in[2];
    const float* W_hh = (const float*)d_in[3];
    const float* b_ih = (const float*)d_in[4];
    const float* b_hh = (const float*)d_in[5];
    const float* W0   = (const float*)d_in[6];
    const float* b0   = (const float*)d_in[7];
    const float* W1   = (const float*)d_in[8];
    const float* b1   = (const float*)d_in[9];
    const float* W2   = (const float*)d_in[10];
    const float* b2   = (const float*)d_in[11];
    const float* W3   = (const float*)d_in[12];
    const float* b3   = (const float*)d_in[13];
    float* out = (float*)d_out;                     // [B, T, OP] fp32

    // workspace layout (~61 MB)
    char* w = (char*)d_ws;
    bf16* Wihb = (bf16*)w; w += (size_t)G4 * I  * sizeof(bf16);
    bf16* Whhb = (bf16*)w; w += (size_t)G4 * Hn * sizeof(bf16);
    bf16* W0b  = (bf16*)w; w += (size_t)D0 * Hn * sizeof(bf16);
    bf16* W1b  = (bf16*)w; w += (size_t)D0 * D0 * sizeof(bf16);
    bf16* W2b  = (bf16*)w; w += (size_t)D1 * D0 * sizeof(bf16);
    bf16* W3b  = (bf16*)w; w += (size_t)OP * D1 * sizeof(bf16);
    bf16* Xseq = (bf16*)w; w += (size_t)B * PAST * I * sizeof(bf16);   // bf16 input copy
    bf16* hbuf = (bf16*)w; w += (size_t)B * Hn * sizeof(bf16);
    bf16* fb   = (bf16*)w; w += (size_t)B * OP * sizeof(bf16);         // bf16 feedback of out_t
    bf16* a0   = (bf16*)w; w += (size_t)B * D0 * sizeof(bf16);
    bf16* a1   = (bf16*)w; w += (size_t)B * D0 * sizeof(bf16);
    bf16* a2   = (bf16*)w; w += (size_t)B * D1 * sizeof(bf16);
    float* cbuf  = (float*)w; w += (size_t)B * Hn * sizeof(float);
    float* gates = (float*)w; w += (size_t)B * G4 * sizeof(float);

    // one-time per launch: bf16 conversions + state zeroing
    auto cvt = [&](const float* s, bf16* d, int n) {
        k_cvt_bf16<<<(n + 255) / 256, 256, 0, stream>>>(s, d, n);
    };
    cvt(W_ih, Wihb, G4 * I);
    cvt(W_hh, Whhb, G4 * Hn);
    cvt(W0,   W0b,  D0 * Hn);
    cvt(W1,   W1b,  D0 * D0);
    cvt(W2,   W2b,  D1 * D0);
    cvt(W3,   W3b,  OP * D1);
    cvt(input_seq, Xseq, B * PAST * I);
    // zero h (bf16, B*Hn*2 bytes) and c (fp32, B*Hn*4 bytes)
    k_zero32<<<(B * Hn / 2) / 256, 256, 0, stream>>>((unsigned*)hbuf, B * Hn / 2);
    k_zero32<<<(B * Hn) / 256, 256, 0, stream>>>((unsigned*)cbuf, B * Hn);

    const dim3 blk(256);
    const dim3 grid_gates(G4 / 64, B / 64);
    const dim3 grid_l0(D0 / 64, B / 64);
    const dim3 grid_l2(D1 / 64, B / 64);
    const dim3 grid_l3(OP / 64, B / 64);

    for (int t = 0; t < T; ++t) {
        const bf16* x;
        int ldx;
        if (t < PAST) { x = Xseq + (size_t)t * I; ldx = PAST * I; }  // [B,PAST,I] slice
        else          { x = fb; ldx = OP; }                          // bf16 feedback

        k_gates<<<grid_gates, blk, 0, stream>>>(x, ldx, hbuf, Wihb, Whhb, b_ih, b_hh, gates);
        k_lstm_pointwise<<<(B * Hn) / 256, blk, 0, stream>>>(gates, hbuf, cbuf);

        k_gemm_bias_relu<<<grid_l0, blk, 0, stream>>>(hbuf, Hn, W0b, Hn, b0, a0, D0,
                                                      (float*)nullptr, 0);
        k_gemm_bias_relu<<<grid_l0, blk, 0, stream>>>(a0, D0, W1b, D0, b1, a1, D0,
                                                      (float*)nullptr, 0);
        k_gemm_bias_relu<<<grid_l2, blk, 0, stream>>>(a1, D0, W2b, D0, b2, a2, D1,
                                                      (float*)nullptr, 0);
        // last layer: bf16 feedback + fp32 output slice
        k_gemm_bias_relu<<<grid_l3, blk, 0, stream>>>(a2, D1, W3b, D1, b3, fb, OP,
                                                      out + (size_t)t * OP, T * OP);
    }
}
